// StandAloneMultiheadAttention_32349693673711
// MI455X (gfx1250) — matrix operand order repaired and checked by reading, not compiled, not run on hardware
//
#include <hip/hip_runtime.h>
#include <hip/hip_bf16.h>

typedef __attribute__((ext_vector_type(16))) _Float16 v16h;
typedef __attribute__((ext_vector_type(8)))  float    v8f;

#define BATCH 2
#define CIN   256
#define HW    56
#define LTOT  3136            // 56*56
#define NH    8
#define HD    32
#define KWIN  49              // 7*7
#define LT16  196             // LTOT/16
#define KT32  8               // CIN/32
#define MAPSZ ((size_t)BATCH * NH * LTOT * HD)   // floats per projection map

// ---------------------------------------------------------------------------
// Pack weights [proj][m=256][c=256] (fp32) into WMMA A-fragment order (f16).
// A tile = 16(M) x 32(K). Lane L holds row m = mt*16 + (L&15); element e:
//   p = e>>1;  k = (p&4?16:0) + (L&16?8:0) + 2*(p&3) + (e&1)
// Tile storage: [proj][mt(16)][kt(8)][lane*16+e]
// ---------------------------------------------------------------------------
__global__ __launch_bounds__(256) void pack_w(const float* __restrict__ Wq,
                                              const float* __restrict__ Wk,
                                              const float* __restrict__ Wv,
                                              _Float16* __restrict__ Apack) {
    int idx = blockIdx.x * blockDim.x + threadIdx.x;      // 3*256*256 total
    int e    = idx & 15;
    int lane = (idx >> 4) & 31;
    int tile = idx >> 9;            // proj*128 + mt*8 + kt
    int kt   = tile & 7;
    int mt   = (tile >> 3) & 15;
    int proj = tile >> 7;
    int p = e >> 1;
    int k = ((p & 4) ? 16 : 0) + ((lane & 16) ? 8 : 0) + 2 * (p & 3) + (e & 1);
    int m = mt * 16 + (lane & 15);
    int c = kt * 32 + k;
    const float* W = (proj == 0) ? Wq : ((proj == 1) ? Wk : Wv);
    Apack[idx] = (_Float16)W[m * CIN + c];
}

// ---------------------------------------------------------------------------
// Pack x [b][c=256][l=3136] (fp32) into WMMA B-fragment order (f16).
// B tile = 32(K) x 16(N). Lane L holds column n = L&15; element e:
//   k = (e&8?16:0) + (L&16?8:0) + (e&7)
// Tile storage: [b][lt(196)][kt(8)][lane*16+e]
// ---------------------------------------------------------------------------
__global__ __launch_bounds__(256) void pack_x(const float* __restrict__ x,
                                              _Float16* __restrict__ Bpack) {
    int idx = blockIdx.x * blockDim.x + threadIdx.x;      // 2*196*8*512 total
    int e    = idx & 15;
    int lane = (idx >> 4) & 31;
    int tile = idx >> 9;            // b*(196*8) + lt*8 + kt
    int kt   = tile & 7;
    int lt   = (tile >> 3) % LT16;
    int b    = tile / (LT16 * 8);
    int k = ((e & 8) ? 16 : 0) + ((lane & 16) ? 8 : 0) + (e & 7);
    int c = kt * 32 + k;
    int l = lt * 16 + (lane & 15);
    Bpack[idx] = (_Float16)x[((size_t)b * CIN + c) * LTOT + l];
}

// ---------------------------------------------------------------------------
// Projection GEMM: out[proj][b][head][l][d] = sum_c W[m=head*32+d][c] * x[c][l]
// One wave per block; 32(M) x 32(L) output block = 2x2 WMMA f32_16x16x32_f16
// fragments; 8 k-steps over CIN=256.
// D fragment: lane L, vgpr v -> row m = 16*i + 8*(L>>4) + v, col n = L&15.
// The 8 vgprs are consecutive d -> store two aligned float4 per subtile.
// ---------------------------------------------------------------------------
__global__ __launch_bounds__(32) void gemm_qkv(const _Float16* __restrict__ Apack,
                                               const _Float16* __restrict__ Bpack,
                                               float* __restrict__ maps) {
    const int lane = threadIdx.x;
    const int lt0  = blockIdx.x * 2;          // two 16-wide column tiles
    const int ybl  = blockIdx.y;              // proj*8 + mblk
    const int proj = ybl >> 3;
    const int head = ybl & 7;                 // m0 = head*32
    const int b    = blockIdx.z;

    const _Float16* Ab = Apack + (size_t)proj * 128 * 512;
    const _Float16* Bb = Bpack + (size_t)b * (LT16 * 8) * 512;
    const int mt0 = head * 2;                 // 16-row A-tile index

    v8f c00 = {}, c01 = {}, c10 = {}, c11 = {};
    #pragma unroll
    for (int kt = 0; kt < KT32; ++kt) {
        v16h a0 = *(const v16h*)(Ab + (((mt0    ) * 8 + kt) * 512) + lane * 16);
        v16h a1 = *(const v16h*)(Ab + (((mt0 + 1) * 8 + kt) * 512) + lane * 16);
        v16h b0 = *(const v16h*)(Bb + (((lt0    ) * 8 + kt) * 512) + lane * 16);
        v16h b1 = *(const v16h*)(Bb + (((lt0 + 1) * 8 + kt) * 512) + lane * 16);
        c00 = __builtin_amdgcn_wmma_f32_16x16x32_f16(false, a0, false, b0, (short)0, c00, false, false);
        c01 = __builtin_amdgcn_wmma_f32_16x16x32_f16(false, a0, false, b1, (short)0, c01, false, false);
        c10 = __builtin_amdgcn_wmma_f32_16x16x32_f16(false, a1, false, b0, (short)0, c10, false, false);
        c11 = __builtin_amdgcn_wmma_f32_16x16x32_f16(false, a1, false, b1, (short)0, c11, false, false);
    }

    float* out = maps + (size_t)proj * MAPSZ;       // [b][head][l][d]
    const int lrow = lane & 15;
    const int hi   = lane >> 4;
    #pragma unroll
    for (int i = 0; i < 2; ++i) {
        #pragma unroll
        for (int j = 0; j < 2; ++j) {
            v8f acc = (i == 0) ? ((j == 0) ? c00 : c01) : ((j == 0) ? c10 : c11);
            int l     = lt0 * 16 + j * 16 + lrow;
            int dbase = i * 16 + hi * 8;
            float* p = out + (((size_t)(b * NH + head) * LTOT + l) * HD + dbase);
            *(float4*)(p)     = make_float4(acc[0], acc[1], acc[2], acc[3]);
            *(float4*)(p + 4) = make_float4(acc[4], acc[5], acc[6], acc[7]);
        }
    }
}

// ---------------------------------------------------------------------------
// Windowed attention with online softmax. One thread per (b, head, l).
// logits[k] = kc . (qp[k] + rel[k]);  OOB cells: qp=vp=0 but rel term remains.
// q/k/v maps are [b][head][l][d] so each vector is 32 contiguous floats.
// ---------------------------------------------------------------------------
__global__ __launch_bounds__(128) void attn_kernel(const float* __restrict__ qmap,
                                                   const float* __restrict__ kmap,
                                                   const float* __restrict__ vmap,
                                                   const float* __restrict__ rel_h,
                                                   const float* __restrict__ rel_w,
                                                   float* __restrict__ out) {
    __shared__ float s_relh[7 * 16];
    __shared__ float s_relw[7 * 16];
    const int n = blockIdx.y;
    const int b = blockIdx.z;
    for (int t = threadIdx.x; t < 224; t += blockDim.x) {
        if (t < 112) s_relh[t] = rel_h[n * 112 + t];
        else         s_relw[t - 112] = rel_w[n * 112 + (t - 112)];
    }
    __syncthreads();

    const int l = blockIdx.x * blockDim.x + threadIdx.x;
    if (l >= LTOT) return;
    const int y = l / HW, xx = l - y * HW;

    const size_t hbase = (size_t)(b * NH + n) * LTOT;
    float kc[HD];
    {
        const float* kp = kmap + (hbase + l) * HD;
        #pragma unroll
        for (int d = 0; d < HD; d += 4) {
            float4 v = *(const float4*)(kp + d);
            kc[d] = v.x; kc[d + 1] = v.y; kc[d + 2] = v.z; kc[d + 3] = v.w;
        }
    }

    float mrun = -INFINITY, srun = 0.0f;
    float acc[HD];
    #pragma unroll
    for (int d = 0; d < HD; ++d) acc[d] = 0.0f;

    #pragma unroll 1
    for (int k = 0; k < KWIN; ++k) {
        const int i = k / 7, j = k - i * 7;
        float s = 0.0f;
        #pragma unroll
        for (int d = 0; d < 16; ++d) s += kc[d]      * s_relh[i * 16 + d];
        #pragma unroll
        for (int d = 0; d < 16; ++d) s += kc[16 + d] * s_relw[j * 16 + d];
        const int qy = y + i - 3, qx = xx + j - 3;
        const bool inb = ((unsigned)qy < (unsigned)HW) && ((unsigned)qx < (unsigned)HW);
        if (inb) {
            const float* qp = qmap + (hbase + (size_t)(qy * HW + qx)) * HD;
            #pragma unroll
            for (int d = 0; d < HD; d += 4) {
                float4 v = *(const float4*)(qp + d);
                s += kc[d] * v.x + kc[d + 1] * v.y + kc[d + 2] * v.z + kc[d + 3] * v.w;
            }
        }
        // online softmax update
        const float mnew = fmaxf(mrun, s);
        const float corr = __expf(mrun - mnew);   // 0 on first iteration
        const float w    = __expf(s - mnew);
        srun = srun * corr + w;
        if (inb) {
            const float* vp = vmap + (hbase + (size_t)(qy * HW + qx)) * HD;
            #pragma unroll
            for (int d = 0; d < HD; d += 4) {
                float4 v = *(const float4*)(vp + d);
                acc[d]     = acc[d]     * corr + w * v.x;
                acc[d + 1] = acc[d + 1] * corr + w * v.y;
                acc[d + 2] = acc[d + 2] * corr + w * v.z;
                acc[d + 3] = acc[d + 3] * corr + w * v.w;
            }
        } else {
            #pragma unroll
            for (int d = 0; d < HD; ++d) acc[d] *= corr;
        }
        mrun = mnew;
    }

    const float inv = 1.0f / srun;
    const size_t ob = ((size_t)b * (NH * HD) + n * HD) * LTOT + l;
    #pragma unroll
    for (int d = 0; d < HD; ++d) out[ob + (size_t)d * LTOT] = acc[d] * inv;
}

// ---------------------------------------------------------------------------
// Launch. Inputs: x, Wk, Wq, Wv, rel_h, rel_w (all fp32). Output fp32.
// Workspace: [qmap|kmap|vmap] fp32 (3 * 6.42MB) + Apack f16 + Bpack f16.
// ---------------------------------------------------------------------------
extern "C" void kernel_launch(void* const* d_in, const int* in_sizes, int n_in,
                              void* d_out, int out_size, void* d_ws, size_t ws_size,
                              hipStream_t stream) {
    const float* x     = (const float*)d_in[0];
    const float* Wk    = (const float*)d_in[1];
    const float* Wq    = (const float*)d_in[2];
    const float* Wv    = (const float*)d_in[3];
    const float* rel_h = (const float*)d_in[4];
    const float* rel_w = (const float*)d_in[5];
    float* out = (float*)d_out;

    char* ws = (char*)d_ws;
    float*    maps  = (float*)ws;                                    // 3 * MAPSZ floats
    _Float16* Apack = (_Float16*)(ws + 3 * MAPSZ * sizeof(float));   // 3*256*256 halfs
    _Float16* Bpack = (_Float16*)(ws + 3 * MAPSZ * sizeof(float) + (size_t)3 * 256 * 256 * 2);

    pack_w<<<(3 * 256 * 256) / 256, 256, 0, stream>>>(Wq, Wk, Wv, Apack);
    pack_x<<<(BATCH * LT16 * 8 * 512) / 256, 256, 0, stream>>>(x, Bpack);
    gemm_qkv<<<dim3(LT16 / 2, 3 * 8, BATCH), 32, 0, stream>>>(Apack, Bpack, maps);
    attn_kernel<<<dim3((LTOT + 127) / 128, NH, BATCH), 128, 0, stream>>>(
        maps, maps + MAPSZ, maps + 2 * MAPSZ, rel_h, rel_w, out);
}
